// MSGSetAbstraction_46145128628930
// MI455X (gfx1250) — matrix operand order repaired and checked by reading, not compiled, not run on hardware
//
#include <hip/hip_runtime.h>
#include <stdint.h>

#ifndef __has_builtin
#define __has_builtin(x) 0
#endif

// ---------------------------------------------------------------------------
// Problem constants (PointNet++ MSG set abstraction)
//   B=8, N=4096, C_IN=64, M=1024, K=64 neighbors, scales out {256,128,128}
// Roofline: MLP = ~60.7 GFLOP vs ~27MB HBM traffic -> compute bound ~20x.
// => bf16 WMMA (f32 accumulate) for all three layers; FPS is a serial
//    latency-bound chain (1 workgroup/batch); ball-query selection fused into
//    the GEMM block and overlapped with the TDM weight fetch into LDS.
// ---------------------------------------------------------------------------
#define B_   8
#define N_   4096
#define CIN_ 64
#define M_   1024
#define KNB  64

typedef __attribute__((ext_vector_type(16))) __bf16   v16bf;
typedef __attribute__((ext_vector_type(8)))  float    v8f;
typedef __attribute__((ext_vector_type(4)))  uint32_t v4u;
typedef __attribute__((ext_vector_type(8)))  int      v8i;
typedef __attribute__((ext_vector_type(4)))  int      v4i;

#if __has_builtin(__builtin_amdgcn_tensor_load_to_lds)
#define HAVE_TDM 1
#else
#define HAVE_TDM 0
#endif

__device__ __forceinline__ unsigned short f32_to_bf16_bits(float x) {
  union { float f; uint32_t u; } c; c.f = x;
  uint32_t r = c.u + 0x7FFFu + ((c.u >> 16) & 1u); // RNE
  return (unsigned short)(r >> 16);
}
__device__ __forceinline__ __bf16 bits_to_bf16(unsigned short v) {
  return __builtin_bit_cast(__bf16, v);
}

// ---------------------------------------------------------------------------
// TDM: 1-D bf16 blob, global -> LDS.  D# per CDNA5 ISA.
// ---------------------------------------------------------------------------
__device__ __forceinline__ void tdm_load_bf16_1d(uint32_t ldsByteOff,
                                                 const void* src,
                                                 uint32_t nElem) {
#if HAVE_TDM
  uint64_t ga = (uint64_t)(uintptr_t)src;
  v4u g0;
  g0[0] = 1u;                                            // count=1, user mode
  g0[1] = ldsByteOff;                                    // lds_addr [63:32]
  g0[2] = (uint32_t)ga;                                  // global_addr lo
  g0[3] = (uint32_t)((ga >> 32) & 0x01FFFFFFu) | (2u << 30); // addr hi | type=2
  v8i g1;
  g1[0] = (int)(1u << 16);                               // data_size=1 (2B), mask=0
  g1[1] = (int)((nElem & 0xFFFFu) << 16);                // tensor_dim0 lo16
  g1[2] = (int)((nElem >> 16) | (1u << 16));             // tensor_dim0 hi | tensor_dim1=1
  g1[3] = (int)((nElem & 0xFFFFu) << 16);                // tile_dim0 = nElem
  g1[4] = 1;                                             // tile_dim1=1, tile_dim2=0
  g1[5] = (int)nElem;                                    // tensor_dim0_stride lo
  g1[6] = 0;
  g1[7] = 0;
  v4i z4 = {0, 0, 0, 0};
#if __has_include(<hip/amd_detail/amd_gfx1250_TDM.h>)
  v8i z8 = {0, 0, 0, 0, 0, 0, 0, 0};
  __builtin_amdgcn_tensor_load_to_lds(g0, g1, z4, z4, z8, 0);   // clang-23 form
#else
  __builtin_amdgcn_tensor_load_to_lds(g0, g1, z4, z4, 0);       // ROCm 7.2 form
#endif
#else
  (void)ldsByteOff; (void)src; (void)nElem;
#endif
}

__device__ __forceinline__ void tdm_wait0() {
#if __has_builtin(__builtin_amdgcn_s_wait_tensorcnt)
  __builtin_amdgcn_s_wait_tensorcnt(0);
#else
  asm volatile("s_wait_tensorcnt 0x0" ::: "memory");
#endif
}

// ---------------------------------------------------------------------------
// WMMA tile loaders, per ISA §7.12.2 VGPR layouts (wave32, 16-bit operands)
//   A 16x32 : lane = {h=lane/16, m=lane%16}; vgpr j holds K = 8h + {2j | 8+2j}
//   B 32x16 : lane = {h=lane/16, n=lane%16}; vgpr j holds K = 8h + {2j | 8+2j}
//   C/D 16x16 f32: lane n = lane%16; vgpr v holds row m = v + 8*(lane/16)
// ---------------------------------------------------------------------------
__device__ __forceinline__ v16bf load_a_tile(const unsigned short* W, int ldk,
                                             int row0, int k0) {
  const int lane = threadIdx.x & 31;
  const int h = lane >> 4;
  const int r = row0 + (lane & 15);
  const unsigned short* base = W + r * ldk + k0 + 8 * h;
  v16bf a;
#pragma unroll
  for (int j = 0; j < 8; ++j) {
    const int kk = (j < 4) ? (2 * j) : (8 + 2 * j); // {0,2,4,6,16,18,20,22}
    a[2 * j]     = bits_to_bf16(base[kk]);
    a[2 * j + 1] = bits_to_bf16(base[kk + 1]);
  }
  return a;
}

__device__ __forceinline__ v16bf load_b_tile(const unsigned short* X,
                                             int k0, int col0) {
  const int lane = threadIdx.x & 31;
  const int h = lane >> 4;
  const int n = col0 + (lane & 15);
  const unsigned short* base = X + (k0 + 8 * h) * 64 + n;
  v16bf b;
#pragma unroll
  for (int j = 0; j < 8; ++j) {
    const int kk = (j < 4) ? (2 * j) : (8 + 2 * j);
    b[2 * j]     = bits_to_bf16(base[kk * 64]);
    b[2 * j + 1] = bits_to_bf16(base[(kk + 1) * 64]);
  }
  return b;
}

// D = relu(W(ROWSxCPAD) @ X(CPADx64) + bias) -> Hout (bf16, row-major [ROWS][64])
template <int ROWS, int CPAD>
__device__ __forceinline__ void mlp_layer_relu(const unsigned short* W,
                                               const float* __restrict__ bias,
                                               const unsigned short* Xin,
                                               unsigned short* Hout) {
  const int wave = threadIdx.x >> 5;
  const int lane = threadIdx.x & 31;
  const int h = lane >> 4;
  constexpr int NT = (ROWS / 16) * 4;
  for (int t = wave; t < NT; t += 8) {
    const int tr = t >> 2, tc = t & 3;
    v8f acc;
#pragma unroll
    for (int v = 0; v < 8; ++v) acc[v] = bias[tr * 16 + v + 8 * h];
#pragma unroll
    for (int ks = 0; ks < CPAD / 32; ++ks) {
      v16bf a  = load_a_tile(W, CPAD, tr * 16, ks * 32);
      v16bf bm = load_b_tile(Xin, ks * 32, tc * 16);
      acc = __builtin_amdgcn_wmma_f32_16x16x32_bf16(false, a, false, bm,
                                                    (short)0, acc, false, false);
    }
    const int n = lane & 15;
#pragma unroll
    for (int v = 0; v < 8; ++v) {
      const float x = fmaxf(acc[v], 0.0f);
      Hout[(tr * 16 + v + 8 * h) * 64 + tc * 16 + n] = f32_to_bf16_bits(x);
    }
  }
}

// out[ch] = max_k( W(ROWSx64) @ X(64x64) )[ch][k] + bias[ch]
template <int ROWS>
__device__ __forceinline__ void mlp_layer_final_max(const unsigned short* W,
                                                    const float* __restrict__ bias,
                                                    const unsigned short* Xin,
                                                    float* __restrict__ outBase) {
  const int wave = threadIdx.x >> 5;
  const int lane = threadIdx.x & 31;
  const int h = lane >> 4;
  for (int tr = wave; tr < ROWS / 16; tr += 8) {
    v8f vmax;
#pragma unroll
    for (int v = 0; v < 8; ++v) vmax[v] = -3.0e38f;
    for (int tc = 0; tc < 4; ++tc) {
      v8f acc;
#pragma unroll
      for (int v = 0; v < 8; ++v) acc[v] = 0.0f;
#pragma unroll
      for (int ks = 0; ks < 2; ++ks) {
        v16bf a  = load_a_tile(W, 64, tr * 16, ks * 32);
        v16bf bm = load_b_tile(Xin, ks * 32, tc * 16);
        acc = __builtin_amdgcn_wmma_f32_16x16x32_bf16(false, a, false, bm,
                                                      (short)0, acc, false, false);
      }
#pragma unroll
      for (int v = 0; v < 8; ++v) vmax[v] = fmaxf(vmax[v], acc[v]);
    }
    // max over the 16 lanes (columns n) that share an output row
#pragma unroll
    for (int msk = 8; msk >= 1; msk >>= 1) {
#pragma unroll
      for (int v = 0; v < 8; ++v)
        vmax[v] = fmaxf(vmax[v], __shfl_xor(vmax[v], msk, 32));
    }
    if ((lane & 15) == 0) {
#pragma unroll
      for (int v = 0; v < 8; ++v) {
        const int ch = tr * 16 + v + 8 * h;
        outBase[(size_t)ch * M_] = vmax[v] + bias[ch];
      }
    }
  }
}

// ---------------------------------------------------------------------------
// Kernel 1: farthest point sampling.  One WG (1024 thr) per batch; the 1023
// argmax steps are a serial chain -> shuffle+LDS two-level reduction per step.
// ---------------------------------------------------------------------------
__global__ void __launch_bounds__(1024)
fps_kernel(const float* __restrict__ pts, float* __restrict__ outCent,
           float* __restrict__ centsWs) {
  const int b = blockIdx.x;
  const int tid = threadIdx.x;
  const float* px = pts + (size_t)b * 3 * N_;
  const float* py = px + N_;
  const float* pz = py + N_;

  float lx[4], ly[4], lz[4], mind[4];
  const float c0x = px[0], c0y = py[0], c0z = pz[0];
#pragma unroll
  for (int i = 0; i < 4; ++i) {
    const int n = tid + i * 1024;
    lx[i] = px[n]; ly[i] = py[n]; lz[i] = pz[n];
    const float dx = lx[i] - c0x, dy = ly[i] - c0y, dz = lz[i] - c0z;
    mind[i] = dx * dx + dy * dy + dz * dz;
  }

  __shared__ float sVal[32];
  __shared__ int   sIdx[32];
  __shared__ float sC[3];

  if (tid == 0) {
    outCent[(size_t)b * 3 * M_ + 0 * M_] = c0x;
    outCent[(size_t)b * 3 * M_ + 1 * M_] = c0y;
    outCent[(size_t)b * 3 * M_ + 2 * M_] = c0z;
    float* cw = centsWs + (size_t)b * M_ * 3;
    cw[0] = c0x; cw[1] = c0y; cw[2] = c0z;
  }

  for (int it = 1; it < M_; ++it) {
    float bv = mind[0];
    int bi = tid;
#pragma unroll
    for (int i = 1; i < 4; ++i) {
      const int n = tid + i * 1024;
      if (mind[i] > bv) { bv = mind[i]; bi = n; }
    }
#pragma unroll
    for (int msk = 16; msk >= 1; msk >>= 1) {
      const float ov = __shfl_xor(bv, msk, 32);
      const int   oi = __shfl_xor(bi, msk, 32);
      if (ov > bv || (ov == bv && oi < bi)) { bv = ov; bi = oi; }
    }
    const int wave = tid >> 5;
    if ((tid & 31) == 0) { sVal[wave] = bv; sIdx[wave] = bi; }
    __syncthreads();
    if (tid < 32) {
      bv = sVal[tid]; bi = sIdx[tid];
#pragma unroll
      for (int msk = 16; msk >= 1; msk >>= 1) {
        const float ov = __shfl_xor(bv, msk, 32);
        const int   oi = __shfl_xor(bi, msk, 32);
        if (ov > bv || (ov == bv && oi < bi)) { bv = ov; bi = oi; }
      }
      if (tid == 0) {
        const float cx = px[bi], cy = py[bi], cz = pz[bi];
        sC[0] = cx; sC[1] = cy; sC[2] = cz;
        outCent[(size_t)b * 3 * M_ + 0 * M_ + it] = cx;
        outCent[(size_t)b * 3 * M_ + 1 * M_ + it] = cy;
        outCent[(size_t)b * 3 * M_ + 2 * M_ + it] = cz;
        float* cw = centsWs + ((size_t)b * M_ + it) * 3;
        cw[0] = cx; cw[1] = cy; cw[2] = cz;
      }
    }
    __syncthreads();
    const float cx = sC[0], cy = sC[1], cz = sC[2];
#pragma unroll
    for (int i = 0; i < 4; ++i) {
      const float dx = lx[i] - cx, dy = ly[i] - cy, dz = lz[i] - cz;
      mind[i] = fminf(mind[i], dx * dx + dy * dy + dz * dz);
    }
  }
}

// ---------------------------------------------------------------------------
// Kernel 2: pack one scale's 3 weight matrices into a padded bf16 blob:
//   [64][96] (layer0, 67 in-ch zero-padded) | [64][64] | [OUT_C][64]
// ---------------------------------------------------------------------------
__global__ void pack_weights_kernel(const float* __restrict__ w0,
                                    const float* __restrict__ w1,
                                    const float* __restrict__ w2,
                                    unsigned short* __restrict__ blob, int outC) {
  const int i = blockIdx.x * blockDim.x + threadIdx.x;
  const int w1e = 64 * 96, w2e = 64 * 64;
  const int tot = w1e + w2e + outC * 64;
  if (i >= tot) return;
  unsigned short v;
  if (i < w1e) {
    const int r = i / 96, c = i % 96;
    v = (c < 67) ? f32_to_bf16_bits(w0[r * 67 + c]) : (unsigned short)0;
  } else if (i < w1e + w2e) {
    v = f32_to_bf16_bits(w1[i - w1e]);
  } else {
    v = f32_to_bf16_bits(w2[i - w1e - w2e]);
  }
  blob[i] = v;
}

// ---------------------------------------------------------------------------
// Kernel 3: fused ball-query (K nearest within radius) + 3-layer bf16-WMMA
// pointnet + max-pool.  One WG (8 waves) per centroid.  Weights stream into
// LDS via TDM while the neighbor selection runs.
// ---------------------------------------------------------------------------
template <int OUT_C>
__global__ void __launch_bounds__(256)
msg_sa_kernel(const float* __restrict__ pts,
              const float* __restrict__ feats,
              const float* __restrict__ cents,
              const unsigned short* __restrict__ wblob,
              const float* __restrict__ b1,
              const float* __restrict__ b2,
              const float* __restrict__ b3,
              float* __restrict__ outFeat,  // (B,512,M) base
              int chanOff, float r2) {
  constexpr int W1E = 64 * 96;
  constexpr int W2E = 64 * 64;
  constexpr int W3E = OUT_C * 64;
  constexpr int WTOT = W1E + W2E + W3E;

  __shared__ unsigned short sW[WTOT];
  __shared__ unsigned short sX[96 * 64];
  __shared__ unsigned short sH[64 * 64];
  __shared__ uint32_t sKey[N_];
  __shared__ uint32_t sPart[8];
  __shared__ int sNbr[KNB];

  const int tid  = threadIdx.x;
  const int wave = tid >> 5;
  const int lane = tid & 31;
  const int bm = blockIdx.x;
  const int b  = bm >> 10;        // / M_
  const int m  = bm & (M_ - 1);

  // --- kick off async weight DMA (overlaps with selection below) ---
#if HAVE_TDM
  if (wave == 0) {
    const uint32_t ldsOff = (uint32_t)(uintptr_t)(&sW[0]);
    tdm_load_bf16_1d(ldsOff, wblob, (uint32_t)WTOT);
  }
#else
  for (int i = tid; i < WTOT; i += 256) sW[i] = wblob[i];
#endif

  const float* px = pts + (size_t)b * 3 * N_;
  const float* py = px + N_;
  const float* pz = py + N_;
  const float cx = cents[(size_t)bm * 3 + 0];
  const float cy = cents[(size_t)bm * 3 + 1];
  const float cz = cents[(size_t)bm * 3 + 2];

  // --- ball query keys: quantized d2 (20b) << 12 | idx (lowest-idx tiebreak)
  for (int i = tid; i < N_; i += 256) {
    const float dx = px[i] - cx, dy = py[i] - cy, dz = pz[i] - cz;
    const float d2 = dx * dx + dy * dy + dz * dz;
    uint32_t key = 0xFFFFFFFFu;
    if (d2 <= r2) {
      const uint32_t q = (uint32_t)(d2 * 4194304.0f);  // 2^22, q < 2^20
      key = (q << 12) | (uint32_t)i;
    }
    sKey[i] = key;
  }
  __syncthreads();

  // --- K rounds of global argmin over 4096 keys (set, not order, matters)
  for (int j = 0; j < KNB; ++j) {
    uint32_t bk = 0xFFFFFFFFu;
    for (int i = tid; i < N_; i += 256) {
      const uint32_t k = sKey[i];
      bk = k < bk ? k : bk;
    }
#pragma unroll
    for (int msk = 16; msk >= 1; msk >>= 1) {
      const uint32_t o = (uint32_t)__shfl_xor((int)bk, msk, 32);
      bk = o < bk ? o : bk;
    }
    if (lane == 0) sPart[wave] = bk;
    __syncthreads();
    if (tid == 0) {
      uint32_t w = sPart[0];
#pragma unroll
      for (int q = 1; q < 8; ++q) w = sPart[q] < w ? sPart[q] : w;
      sNbr[j] = (w == 0xFFFFFFFFu) ? -1 : (int)(w & 0xFFFu);
      if (w != 0xFFFFFFFFu) sKey[w & 0xFFFu] = 0xFFFFFFFFu;
    }
    __syncthreads();
  }

  // --- stage grouped input X: rows 0..2 = rel xyz, 3..66 = features, pad->96
  for (int i = tid; i < 96 * 64; i += 256) sX[i] = 0;
  __syncthreads();
  {
    const int j = tid >> 2;     // neighbor slot 0..63
    const int part = tid & 3;   // 16 feature channels each
    const int n = sNbr[j];
    if (n >= 0) {
      if (part == 0) {
        sX[0 * 64 + j] = f32_to_bf16_bits(px[n] - cx);
        sX[1 * 64 + j] = f32_to_bf16_bits(py[n] - cy);
        sX[2 * 64 + j] = f32_to_bf16_bits(pz[n] - cz);
      }
      const float* fb = feats + (size_t)b * CIN_ * N_ + (size_t)(part * 16) * N_ + n;
#pragma unroll
      for (int c = 0; c < 16; ++c)
        sX[(3 + part * 16 + c) * 64 + j] = f32_to_bf16_bits(fb[(size_t)c * N_]);
    }
  }
#if HAVE_TDM
  if (wave == 0) tdm_wait0();
#endif
  __syncthreads();

  // --- 3-layer pointnet with bf16 WMMA + fused max-pool over K ---
  mlp_layer_relu<64, 96>(sW, b1, sX, sH);
  __syncthreads();
  mlp_layer_relu<64, 64>(sW + W1E, b2, sH, sX);   // reuse sX[0..4095] as H2
  __syncthreads();
  float* outBase = outFeat + ((size_t)b * 512 + chanOff) * M_ + m;
  mlp_layer_final_max<OUT_C>(sW + W1E + W2E, b3, sX, outBase);
}

// ---------------------------------------------------------------------------
// Host launcher
// ---------------------------------------------------------------------------
extern "C" void kernel_launch(void* const* d_in, const int* in_sizes, int n_in,
                              void* d_out, int out_size, void* d_ws, size_t ws_size,
                              hipStream_t stream) {
  (void)out_size; (void)ws_size;

  const float* points = nullptr;
  const float* feats  = nullptr;
  const float* rest[18];
  int restSz[18];
  int nr = 0;
  for (int i = 0; i < n_in; ++i) {
    const int sz = in_sizes[i];
    if (sz == B_ * 3 * N_ && !points) points = (const float*)d_in[i];
    else if (sz == B_ * CIN_ * N_ && !feats) feats = (const float*)d_in[i];
    else if (nr < 18) { rest[nr] = (const float*)d_in[i]; restSz[nr] = sz; ++nr; }
  }
  if (!points || !feats || nr < 18) return;

  // params pytree flattens as either all-w-then-all-b (insertion order) or
  // all-b-then-all-w (sorted keys); each group is [s0:{l0,l1,l2}, s1, s2].
  const float *W[9], *Bs[9];
  const bool wFirst = (restSz[0] > 1000);
  for (int k = 0; k < 9; ++k) {
    W[k]  = wFirst ? rest[k]     : rest[9 + k];
    Bs[k] = wFirst ? rest[9 + k] : rest[k];
  }

  // workspace layout
  float* centsWs = (float*)d_ws;                                   // B*M*3 f32
  unsigned short* blob0 =
      (unsigned short*)((char*)d_ws + (size_t)B_ * M_ * 3 * sizeof(float));
  const int tot0  = 64 * 96 + 64 * 64 + 256 * 64;
  const int tot12 = 64 * 96 + 64 * 64 + 128 * 64;
  unsigned short* blob1 = blob0 + tot0;
  unsigned short* blob2 = blob1 + tot12;

  pack_weights_kernel<<<(tot0  + 255) / 256, 256, 0, stream>>>(W[0], W[1], W[2], blob0, 256);
  pack_weights_kernel<<<(tot12 + 255) / 256, 256, 0, stream>>>(W[3], W[4], W[5], blob1, 128);
  pack_weights_kernel<<<(tot12 + 255) / 256, 256, 0, stream>>>(W[6], W[7], W[8], blob2, 128);

  float* outCent = (float*)d_out;                 // (B,3,M)
  float* outFeat = outCent + (size_t)B_ * 3 * M_; // (B,512,M)

  fps_kernel<<<B_, 1024, 0, stream>>>(points, outCent, centsWs);

  msg_sa_kernel<256><<<B_ * M_, 256, 0, stream>>>(points, feats, centsWs, blob0,
                                                  Bs[0], Bs[1], Bs[2], outFeat,
                                                  0, 0.01f);
  msg_sa_kernel<128><<<B_ * M_, 256, 0, stream>>>(points, feats, centsWs, blob1,
                                                  Bs[3], Bs[4], Bs[5], outFeat,
                                                  256, 0.04f);
  msg_sa_kernel<128><<<B_ * M_, 256, 0, stream>>>(points, feats, centsWs, blob2,
                                                  Bs[6], Bs[7], Bs[8], outFeat,
                                                  384, 0.16f);
}
